// PGJANET_27023934227289
// MI455X (gfx1250) — matrix operand order repaired and checked by reading, not compiled, not run on hardware
//
#include <hip/hip_runtime.h>
#include <math.h>

typedef _Float16 v16h __attribute__((ext_vector_type(16)));
typedef _Float16 v8h  __attribute__((ext_vector_type(8)));
typedef float    v8f  __attribute__((ext_vector_type(8)));
typedef float    v4f  __attribute__((ext_vector_type(4)));

#define B_   256
#define T_   2048
#define H_   64
#define O_   2
#define MT   16            // batch rows per workgroup (one WMMA M-tile)
#define WAVES 12
#define NTHR (WAVES * 32)

// ---- fast, branchless transcendentals --------------------------------------

__device__ __forceinline__ float fast_rcp(float x) {
  return __builtin_amdgcn_rcpf(x);                 // v_rcp_f32
}

__device__ __forceinline__ float fast_exp2(float x) {
#if __has_builtin(__builtin_amdgcn_exp2f)
  return __builtin_amdgcn_exp2f(x);                // v_exp_f32
#else
  return exp2f(x);
#endif
}

__device__ __forceinline__ float fast_tanh(float x) {
#if __has_builtin(__builtin_amdgcn_tanhf)
  return __builtin_amdgcn_tanhf(x);                // v_tanh_f32 (gfx1250 tanh-insts)
#else
  float xc = fminf(fmaxf(x, -10.0f), 10.0f);
  float e  = fast_exp2(xc * 2.8853900817779268f);  // exp(2x)
  return (e - 1.0f) * fast_rcp(e + 1.0f);
#endif
}

__device__ __forceinline__ float fast_sigmoid(float x) {
  return fast_rcp(1.0f + fast_exp2(x * -1.4426950408889634f));
}

__device__ __forceinline__ float fast_rsq(float x) {
#if __has_builtin(__builtin_amdgcn_rsqf)
  return __builtin_amdgcn_rsqf(x);                 // v_rsq_f32
#else
  return rsqrtf(x);
#endif
}

// ---- WMMA helpers -----------------------------------------------------------

__device__ __forceinline__ v8f wmma_f16(v16h a, v16h b, v8f c) {
  return __builtin_amdgcn_wmma_f32_16x16x32_f16(
      false, a, false, b, (short)0, c, false, false);
}

// A-matrix fragment (16x32 f16) from an LDS row-major [16 x 64] f16 tile.
__device__ __forceinline__ v16h load_a16(const _Float16* src, int kc, int lane) {
  int m  = lane & 15;
  int k0 = kc + ((lane >> 4) << 3);
  const _Float16* p = src + m * H_ + k0;
  v8h lo = *(const v8h*)(p);
  v8h hi = *(const v8h*)(p + 16);
  v16h r;
#pragma unroll
  for (int i = 0; i < 8; ++i) { r[i] = lo[i]; r[8 + i] = hi[i]; }
  return r;
}

// B-matrix fragment (32x16 f16) from a global row-major f32 weight (n_out x rs).
__device__ __forceinline__ v16h load_b16(const float* W, int rs, int n, int kc,
                                         int nvalid, int lane) {
  int k0 = kc + ((lane >> 4) << 3);
  v16h r;
#pragma unroll
  for (int i = 0; i < 16; ++i) {
    float v = (n < nvalid) ? W[n * rs + k0 + i + ((i >> 3) << 3)] : 0.0f;
    r[i] = (_Float16)v;
  }
  return r;
}

// ---- kernel -----------------------------------------------------------------

__global__ __launch_bounds__(NTHR) void pgjanet_kernel(
    const float* __restrict__ x,  const float* __restrict__ h0,
    const float* __restrict__ Wa, const float* __restrict__ ba,
    const float* __restrict__ Wp1, const float* __restrict__ bp1,
    const float* __restrict__ Wp2, const float* __restrict__ bp2,
    const float* __restrict__ Wf, const float* __restrict__ bf,
    const float* __restrict__ Wg, const float* __restrict__ bg,
    const float* __restrict__ Wo, const float* __restrict__ bo,
    float* __restrict__ out)
{
  __shared__ __align__(32) _Float16 sh_h16[MT * H_];   // h_t as f16 (WMMA A)
  __shared__ __align__(32) _Float16 sh_u16[MT * H_];   // u_t as f16 (WMMA A)
  // transposed staging [n*16+m] -> contiguous v8f per phase-1 lane
  __shared__ __align__(32) float    sh_aT [H_ * MT];
  __shared__ __align__(32) float    sh_p1T[H_ * MT];
  __shared__ __align__(32) float    sh_p2T[H_ * MT];
  __shared__ __align__(32) float sh_amp[MT], sh_cos[MT], sh_sin[MT];

  const int tid   = threadIdx.x;
  const int wave  = tid >> 5;
  const int lane  = tid & 31;
  const int bbase = blockIdx.x * MT;
  const int mofs  = (lane >> 4) << 3;   // C/D layout: lanes>=16 cover M+8

  // ---- init h16 from h0 -----------------------------------------------------
  for (int idx = tid; idx < MT * H_; idx += NTHR) {
    int m = idx >> 6, n = idx & 63;
    sh_h16[idx] = (_Float16)h0[(size_t)(bbase + m) * H_ + n];
  }

  // ---- preload loop-invariant B fragments / scalars into registers ----------
  // Phase-1 task: wave -> (matrix = wave/4 in {a,p1,p2}, N-tile = wave%4)
  const int   p1mat  = wave >> 2;
  const int   p1tile = wave & 3;
  const float* W1 = (p1mat == 0) ? Wa : ((p1mat == 1) ? Wp1 : Wp2);
  const float* b1 = (p1mat == 0) ? ba : ((p1mat == 1) ? bp1 : bp2);
  const int   n1 = p1tile * 16 + (lane & 15);
  v16h B1k0 = load_b16(W1, H_ + 1, n1, 0,  H_, lane);
  v16h B1k1 = load_b16(W1, H_ + 1, n1, 32, H_, lane);
  const float bias1  = b1[n1];
  const float wlast1 = W1[n1 * (H_ + 1) + H_];   // rank-1 column for concat input

  // Phase-2 task: waves 0-3 -> N-tile = wave, both f and g.
  // f32 hidden state lives in registers: hreg[i] = h[(i+mofs), n2].
  v16h Bfr[4], Bgr[4];
  v8f  hreg = {};
  float biasf = 0.0f, biasg = 0.0f;
  const int n2 = (wave < 4) ? (wave * 16 + (lane & 15)) : 0;
  if (wave < 4) {
#pragma unroll
    for (int c = 0; c < 4; ++c) {
      Bfr[c] = load_b16(Wf, 2 * H_, n2, 32 * c, H_, lane);
      Bgr[c] = load_b16(Wg, 2 * H_, n2, 32 * c, H_, lane);
    }
    biasf = bf[n2];
    biasg = bg[n2];
#pragma unroll
    for (int i = 0; i < 8; ++i)
      hreg[i] = h0[(size_t)(bbase + i + mofs) * H_ + n2];
  }

  // Output task: wave 4, O padded 2 -> 16
  v16h By0 = {}, By1 = {};
  float biaso = 0.0f;
  const int ny = lane & 15;
  if (wave == 4) {
    By0 = load_b16(Wo, H_, ny, 0,  O_, lane);
    By1 = load_b16(Wo, H_, ny, 32, O_, lane);
    biaso = (ny < O_) ? bo[ny] : 0.0f;
  }
  __syncthreads();

  // ---- recurrence -----------------------------------------------------------
  for (int t = 0; t < T_; ++t) {
    // Prep: amp, cos=i*rsq, sin=q*rsq (branchless, no trig)
    if (tid < MT) {
      const size_t xo = ((size_t)(bbase + tid) * T_ + t) * 2;
      float iv = x[xo + 0];
      float qv = x[xo + 1];
      float s2 = iv * iv + qv * qv;
      bool  nz = s2 > 0.0f;
      float inv = nz ? fast_rsq(s2) : 0.0f;
      sh_amp[tid] = s2 * inv;
      sh_cos[tid] = nz ? iv * inv : 1.0f;          // atan2(0,0)=0
      sh_sin[tid] = qv * inv;
      if (t + 1 < T_) __builtin_prefetch(&x[xo + 2], 0, 3);
    }
    __syncthreads();

    // Phase 1: a / p1 / p2  (12 waves, 2 WMMAs each, K = 64 over h)
    v16h ah0 = load_a16(sh_h16, 0,  lane);
    v16h ah1 = load_a16(sh_h16, 32, lane);
    {
      v8f acc = {};
      acc = wmma_f16(ah0, B1k0, acc);
      acc = wmma_f16(ah1, B1k1, acc);
      const float* ext   = (p1mat == 0) ? sh_amp : ((p1mat == 1) ? sh_cos : sh_sin);
      float*       stage = (p1mat == 0) ? sh_aT : ((p1mat == 1) ? sh_p1T : sh_p2T);
      v8f ev = *(const v8f*)(ext + mofs);          // vector preload: 8 independent fma+tanh
      v8f res;
#pragma unroll
      for (int i = 0; i < 8; ++i)
        res[i] = fast_tanh(acc[i] + bias1 + ev[i] * wlast1);
      *(v8f*)(stage + n1 * MT + mofs) = res;       // contiguous transposed store
    }
    __syncthreads();

    // u = a*p1*p2*(1-a)*(1-p1)*(1-p2)  -- 256 threads x 4 contiguous elements
    if (tid < 256) {
      const int base = tid * 4;                    // idx = n*16+m, m = base&15..+3
      const int n = base >> 4;
      const int m = base & 15;
      v4f av = *(const v4f*)(sh_aT  + base);
      v4f p1v = *(const v4f*)(sh_p1T + base);
      v4f p2v = *(const v4f*)(sh_p2T + base);
#pragma unroll
      for (int i = 0; i < 4; ++i) {
        float u = av[i] * p1v[i] * p2v[i] *
                  (1.0f - av[i]) * (1.0f - p1v[i]) * (1.0f - p2v[i]);
        sh_u16[(m + i) * H_ + n] = (_Float16)u;
      }
    }
    __syncthreads();

    // Phase 2 WMMA: f, g over [h | u]  (waves 0-3, 8 WMMAs each)
    v8f accf = {}, accg = {};
    if (wave < 4) {
      v16h au0 = load_a16(sh_u16, 0,  lane);
      v16h au1 = load_a16(sh_u16, 32, lane);
      accf = wmma_f16(ah0, Bfr[0], accf);
      accf = wmma_f16(ah1, Bfr[1], accf);
      accf = wmma_f16(au0, Bfr[2], accf);
      accf = wmma_f16(au1, Bfr[3], accf);
      accg = wmma_f16(ah0, Bgr[0], accg);
      accg = wmma_f16(ah1, Bgr[1], accg);
      accg = wmma_f16(au0, Bgr[2], accg);
      accg = wmma_f16(au1, Bgr[3], accg);
    }
    __syncthreads();   // all phase-2 A-fragment reads of h done before h update

    // Phase 2 epilogue: h_new = sigmoid(f)*h + (1-sigmoid(f))*tanh(g)
    // h stays in registers; only the f16 copy for WMMA goes to LDS.
    if (wave < 4) {
#pragma unroll
      for (int i = 0; i < 8; ++i) {
        float fv = fast_sigmoid(accf[i] + biasf);
        float gv = fast_tanh(accg[i] + biasg);
        float hn = fv * hreg[i] + (1.0f - fv) * gv;
        hreg[i] = hn;
        sh_h16[(i + mofs) * H_ + n2] = (_Float16)hn;
      }
    }
    __syncthreads();

    // Output y_t = h_new @ Wo^T + bo  (wave 4; overlaps next step's prep)
    if (wave == 4) {
      v16h an0 = load_a16(sh_h16, 0,  lane);
      v16h an1 = load_a16(sh_h16, 32, lane);
      v8f acy = {};
      acy = wmma_f16(an0, By0, acy);
      acy = wmma_f16(an1, By1, acy);
      if (ny < O_) {
#pragma unroll
        for (int i = 0; i < 8; ++i) {
          int m = i + mofs;
          out[((size_t)(bbase + m) * T_ + t) * O_ + ny] = acy[i] + biaso;
        }
      }
    }
  }
}

// ---- host launch ------------------------------------------------------------

extern "C" void kernel_launch(void* const* d_in, const int* in_sizes, int n_in,
                              void* d_out, int out_size, void* d_ws, size_t ws_size,
                              hipStream_t stream) {
  const float* x   = (const float*)d_in[0];
  const float* h0  = (const float*)d_in[1];
  const float* Wa  = (const float*)d_in[2];
  const float* ba  = (const float*)d_in[3];
  const float* Wp1 = (const float*)d_in[4];
  const float* bp1 = (const float*)d_in[5];
  const float* Wp2 = (const float*)d_in[6];
  const float* bp2 = (const float*)d_in[7];
  const float* Wf  = (const float*)d_in[8];
  const float* bf  = (const float*)d_in[9];
  const float* Wg  = (const float*)d_in[10];
  const float* bg  = (const float*)d_in[11];
  const float* Wo  = (const float*)d_in[12];
  const float* bo  = (const float*)d_in[13];
  float* out = (float*)d_out;

  dim3 grid(B_ / MT);     // 16 workgroups, one 16-row batch tile each
  dim3 block(NTHR);       // 12 waves (wave32)
  pgjanet_kernel<<<grid, block, 0, stream>>>(x, h0, Wa, ba, Wp1, bp1, Wp2, bp2,
                                             Wf, bf, Wg, bg, Wo, bo, out);
}
